// SoftUnifAttention_61340722921964
// MI455X (gfx1250) — matrix operand order repaired and checked by reading, not compiled, not run on hardware
//
#include <hip/hip_runtime.h>

// ---------------------------------------------------------------------------
// SoftUnifAttention forward for MI455X (gfx1250, wave32, WMMA bf16, TDM)
// B=2 S=2048 H=1024 NH=16 HD=64 NT=5
// d_out layout: [output 2*2048*1024][type_logits 2*2048*5][unif_logits 2*16*2048*2048]
// ---------------------------------------------------------------------------

#define CB  2
#define CS  2048
#define CH  1024
#define CNH 16
#define CHD 64
#define CNT 5

#define USE_TDM 1

typedef __attribute__((ext_vector_type(8)))  float  v8f;
typedef __attribute__((ext_vector_type(8)))  __bf16 v8bf;
typedef __attribute__((ext_vector_type(16))) __bf16 v16bf;
typedef __attribute__((ext_vector_type(4)))  unsigned u32x4;
typedef __attribute__((ext_vector_type(8)))  unsigned u32x8;

__device__ __forceinline__ __bf16 f2bf(float f) {
    union { float f; unsigned u; } v; v.f = f;
    unsigned r = v.u + 0x7fffu + ((v.u >> 16) & 1u);   // round-to-nearest-even
    unsigned short h = (unsigned short)(r >> 16);
    return __builtin_bit_cast(__bf16, h);
}

#if USE_TDM
// Issue one TDM 2D tile load: 64 rows x 64 bf16, row stride `row_stride_elems`,
// into LDS at byte address `lds_addr` with 80-bf16 padded rows
// (pad_interval=4 -> pad after every 32 DWORDs; pad_amount=7 -> 8 DWORDs = 32B).
__device__ __forceinline__ void tdm_load_tile64(unsigned lds_addr, const void* gptr,
                                                unsigned row_stride_elems) {
    size_t ga = (size_t)gptr;
    u32x4 g0 = { 1u,                                   // count=1 (valid user D#)
                 lds_addr,                             // LDS byte address
                 (unsigned)ga,                         // global_addr[31:0]
                 (unsigned)((ga >> 32) & 0x01FFFFFFu) | 0x80000000u }; // [56:32], type=2
    u32x8 g1 = { (1u << 16) | (1u << 20) | (4u << 22) | (7u << 25),
                 //  data_size=2B | pad_enable | pad_interval=32dw | pad_amount=8dw
                 (64u << 16),                          // tensor_dim0[15:0]=64 (abar=0)
                 (64u << 16),                          // tensor_dim1[15:0]=64 (td0 hi=0)
                 (64u << 16),                          // tile_dim0=64 (td1 hi=0)
                 64u,                                  // tile_dim1=64, tile_dim2=0
                 row_stride_elems,                     // tensor_dim0_stride[31:0]
                 0u, 0u };
    u32x4 gz = { 0u, 0u, 0u, 0u };
    asm volatile("tensor_load_to_lds %0, %1, %2, %3"
                 :: "s"(g0), "s"(g1), "s"(gz), "s"(gz)
                 : "memory");
}
#endif

// ------------------------------- f32 -> bf16 -------------------------------
__global__ void __launch_bounds__(256)
cvt_kernel(const float* __restrict__ in, __bf16* __restrict__ out, int n) {
    int i = blockIdx.x * blockDim.x + threadIdx.x;
    int stride = gridDim.x * blockDim.x;
    for (; i < n; i += stride) out[i] = f2bf(in[i]);
}

// ---------------- V transpose: [B,S,NH*HD] -> [B,NH,HD,S] ------------------
__global__ void __launch_bounds__(256)
vtrans_kernel(const __bf16* __restrict__ v, __bf16* __restrict__ vt) {
    __shared__ __bf16 t[64 * 72];
    const int s0 = blockIdx.x * 64, h = blockIdx.y, b = blockIdx.z;
    const int tid = threadIdx.x;
    #pragma unroll
    for (int j = 0; j < 2; ++j) {
        int idx8 = tid + j * 256;
        int s = idx8 >> 3, d = (idx8 & 7) * 8;
        v8bf vv = *(const v8bf*)&v[((size_t)(b * CS + s0 + s)) * CH + h * CHD + d];
        #pragma unroll
        for (int i = 0; i < 8; ++i) t[(d + i) * 72 + s] = vv[i];
    }
    __syncthreads();
    #pragma unroll
    for (int j = 0; j < 2; ++j) {
        int idx8 = tid + j * 256;
        int d = idx8 >> 3, s = (idx8 & 7) * 8;
        *(v8bf*)&vt[(((size_t)b * CNH + h) * CHD + d) * CS + s0 + s] =
            *(const v8bf*)&t[d * 72 + s];
    }
}

// ---------------- type inference: logits, softmax, latent, qt, kt ----------
__global__ void __launch_bounds__(128)
type_kernel(const float* __restrict__ x,
            const float* __restrict__ Wti, const float* __restrict__ bti,
            const float* __restrict__ temb,
            const float* __restrict__ Wqt, const float* __restrict__ bqt,
            const float* __restrict__ Wkt, const float* __restrict__ bkt,
            float* __restrict__ out_logits,
            float* __restrict__ qt_f32, __bf16* __restrict__ ktb) {
    const int row = blockIdx.x;            // b*S+s
    const int tid = threadIdx.x;           // 128 threads
    __shared__ float red[CNT][128];
    __shared__ float lg[CNT];
    __shared__ float lat[CHD];
    const float* xr = x + (size_t)row * CH;
    for (int t = 0; t < CNT; ++t) {
        float p = 0.f;
        for (int i = tid; i < CH; i += 128) p += xr[i] * Wti[t * CH + i];
        red[t][tid] = p;
    }
    __syncthreads();
    if (tid < CNT) {
        float s = 0.f;
        for (int i = 0; i < 128; ++i) s += red[tid][i];
        s += bti[tid];
        lg[tid] = s;
        out_logits[(size_t)row * CNT + tid] = s;
    }
    __syncthreads();
    if (tid < CHD) {
        float mx = lg[0];
        for (int t = 1; t < CNT; ++t) mx = fmaxf(mx, lg[t]);
        float den = 0.f, p[CNT];
        for (int t = 0; t < CNT; ++t) { p[t] = __expf(lg[t] - mx); den += p[t]; }
        float iden = __builtin_amdgcn_rcpf(den);
        float l = 0.f;
        for (int t = 0; t < CNT; ++t) l += (p[t] * iden) * temb[t * CHD + tid];
        lat[tid] = l;
    }
    __syncthreads();
    if (tid < CHD) {
        float q = bqt[tid], k = bkt[tid];
        for (int e = 0; e < CHD; ++e) {
            float le = lat[e];
            q += le * Wqt[tid * CHD + e];
            k += le * Wkt[tid * CHD + e];
        }
        qt_f32[(size_t)row * CHD + tid] = q;
        ktb[(size_t)row * CHD + tid] = f2bf(k);
    }
}

// ---------------- qtb[b,h,s,e] = sum_d qt[b,s,d] * W_bil[h,d,e] ------------
__global__ void __launch_bounds__(64)
qtbil_kernel(const float* __restrict__ qt, const float* __restrict__ Wbil,
             __bf16* __restrict__ qtb) {
    const int row = blockIdx.x;            // b*S+s
    const int h = blockIdx.y;
    const int e = threadIdx.x;             // 64 threads
    __shared__ float lq[CHD];
    lq[e] = qt[(size_t)row * CHD + e];
    __syncthreads();
    const float* Wb = Wbil + (size_t)h * CHD * CHD;
    float acc = 0.f;
    for (int d = 0; d < CHD; ++d) acc += lq[d] * Wb[d * CHD + e];
    const int b = row / CS, s = row % CS;
    qtb[((((size_t)b * CNH + h) * CS) + s) * CHD + e] = f2bf(acc);
}

// ---------------- WMMA GEMM: C[M,N] = A[M,K] @ B[N,K]^T + bias -------------
template <bool BF16_OUT>
__global__ void __launch_bounds__(256, 2)
gemm_bf16(const __bf16* __restrict__ A, const __bf16* __restrict__ Bw,
          const float* __restrict__ bias, void* __restrict__ Cout,
          int M, int N, int K) {
    constexpr int LDT = 40;                        // 32 + 8 pad (bf16)
    __shared__ __attribute__((aligned(32))) __bf16 sA[128 * LDT];
    __shared__ __attribute__((aligned(32))) __bf16 sB[128 * LDT];
    const int tid = threadIdx.x;
    const int wave = tid >> 5, lane = tid & 31;
    const int l16 = lane & 15, g = lane >> 4;
    const int mbase = blockIdx.y * 128;
    const int nbase = blockIdx.x * 128;
    const int w16 = wave * 16;

    v8f acc[8];
    #pragma unroll
    for (int i = 0; i < 8; ++i)
        #pragma unroll
        for (int r = 0; r < 8; ++r) acc[i][r] = 0.f;

    #pragma unroll 1
    for (int k0 = 0; k0 < K; k0 += 32) {
        __syncthreads();
        #pragma unroll
        for (int j = 0; j < 2; ++j) {
            int idx8 = tid + j * 256;              // 512 v8 loads per tile
            int m = idx8 >> 2;
            int kk = (idx8 & 3) * 8;
            *(v8bf*)&sA[m * LDT + kk] = *(const v8bf*)&A[(size_t)(mbase + m) * K + k0 + kk];
            *(v8bf*)&sB[m * LDT + kk] = *(const v8bf*)&Bw[(size_t)(nbase + m) * K + k0 + kk];
        }
        __syncthreads();
        if (k0 + 32 < K) {
            __builtin_prefetch(&A[(size_t)(mbase + (tid >> 1)) * K + k0 + 32], 0, 1);
            __builtin_prefetch(&Bw[(size_t)(nbase + (tid >> 1)) * K + k0 + 32], 0, 1);
        }
        const __bf16* arow = &sA[(w16 + l16) * LDT];
        v16bf afrag;
        ((v8bf*)&afrag)[0] = *(const v8bf*)(arow + g * 8);
        ((v8bf*)&afrag)[1] = *(const v8bf*)(arow + 16 + g * 8);
        #pragma unroll
        for (int nt = 0; nt < 8; ++nt) {
            const __bf16* brow = &sB[(nt * 16 + l16) * LDT];
            v16bf bfrag;
            ((v8bf*)&bfrag)[0] = *(const v8bf*)(brow + g * 8);
            ((v8bf*)&bfrag)[1] = *(const v8bf*)(brow + 16 + g * 8);
            acc[nt] = __builtin_amdgcn_wmma_f32_16x16x32_bf16(
                false, afrag, false, bfrag, (short)0, acc[nt], false, false);
        }
    }
    if (BF16_OUT) {
        __bf16* Cp = (__bf16*)Cout + (size_t)(mbase + w16 + g * 8) * N + nbase + l16;
        #pragma unroll
        for (int nt = 0; nt < 8; ++nt) {
            float bb = bias[nbase + nt * 16 + l16];
            #pragma unroll
            for (int r = 0; r < 8; ++r)
                Cp[r * N + nt * 16] = f2bf(acc[nt][r] + bb);
        }
    } else {
        float* Cp = (float*)Cout + (size_t)(mbase + w16 + g * 8) * N + nbase + l16;
        #pragma unroll
        for (int nt = 0; nt < 8; ++nt) {
            float bb = bias[nbase + nt * 16 + l16];
            #pragma unroll
            for (int r = 0; r < 8; ++r)
                Cp[r * N + nt * 16] = acc[nt][r] + bb;
        }
    }
}

// ---------------- fused flash attention + bilinear unification -------------
// grid (S/128, NH, B), 256 threads (8 waves x 16 q-rows). k-blocks of 64.
// vt is V pre-transposed: [B,NH,HD,S].
// Softmax runs in log2 domain: comb2 = cv*(scale*log2e) + log2(sig+eps) + madd2,
// p = exp2(comb2 - rowmax2)  (identical values to exp-domain softmax).
__global__ void __launch_bounds__(256, 1)
flash_kernel(const __bf16* __restrict__ qb, const __bf16* __restrict__ kb,
             const __bf16* __restrict__ vt, const __bf16* __restrict__ qtbb,
             const __bf16* __restrict__ ktb, const int* __restrict__ mask,
             float* __restrict__ unif_out, float* __restrict__ aout) {
    constexpr int LD = 80;                         // 64 + 16 pad (bf16), 32B-aligned rows
    constexpr int TILE = 64 * LD;                  // 5120 elems per tile
    constexpr int BUFE = 3 * TILE;                 // K,KT,VT set = 15360 elems
    // [buf0: 3 tiles][buf1: 3 tiles][sP: 8*16*LD]  (sQ staging aliases sP)
    __shared__ __attribute__((aligned(32))) __bf16 smem[2 * BUFE + 8 * 16 * LD];
    __shared__ float sMadd[64];
    __bf16* sP = smem + 2 * BUFE;                  // 10240 elems
    __bf16* sQ = sP;                               // prologue-only alias

    const int tid = threadIdx.x;
    const int wave = tid >> 5, lane = tid & 31;
    const int l16 = lane & 15, g = lane >> 4;
    const int qbase = blockIdx.x * 128;
    const int h = blockIdx.y, b = blockIdx.z;
    const int w16 = wave * 16;
    const float scale = 0.125f;                    // 1/sqrt(64)
    const float sc2 = 0.18033688f;                 // scale * log2(e)
    __bf16* myP = sP + wave * 16 * LD;

    // ---- stage Q head-slice, gather loop-invariant A fragments ----
    #pragma unroll
    for (int j = 0; j < 4; ++j) {
        int idx8 = tid + j * 256;                  // 128x64 -> 1024 v8 loads
        int m = idx8 >> 3, d = (idx8 & 7) * 8;
        *(v8bf*)&sQ[m * LD + d] =
            *(const v8bf*)&qb[((size_t)(b * CS + qbase + m)) * CH + h * CHD + d];
    }
    __syncthreads();
    v16bf qfrag[2], tfrag[2];
    {
        const __bf16* arow = &sQ[(w16 + l16) * LD];
        #pragma unroll
        for (int ds = 0; ds < 2; ++ds) {
            ((v8bf*)&qfrag[ds])[0] = *(const v8bf*)(arow + ds * 32 + g * 8);
            ((v8bf*)&qfrag[ds])[1] = *(const v8bf*)(arow + ds * 32 + 16 + g * 8);
        }
    }
    __syncthreads();
    #pragma unroll
    for (int j = 0; j < 4; ++j) {
        int idx8 = tid + j * 256;
        int m = idx8 >> 3, d = (idx8 & 7) * 8;
        *(v8bf*)&sQ[m * LD + d] =
            *(const v8bf*)&qtbb[((((size_t)b * CNH + h) * CS) + qbase + m) * CHD + d];
    }
    __syncthreads();
    {
        const __bf16* arow = &sQ[(w16 + l16) * LD];
        #pragma unroll
        for (int ds = 0; ds < 2; ++ds) {
            ((v8bf*)&tfrag[ds])[0] = *(const v8bf*)(arow + ds * 32 + g * 8);
            ((v8bf*)&tfrag[ds])[1] = *(const v8bf*)(arow + ds * 32 + 16 + g * 8);
        }
    }
    __syncthreads();                               // LDS free for tile buffers now

    const __bf16* kG  = &kb[(size_t)(b * CS) * CH + h * CHD];
    const __bf16* ktG = &ktb[(size_t)(b * CS) * CHD];
    const __bf16* vtG = &vt[(((size_t)b * CNH + h) * CHD) * CS];

#if USE_TDM
    if (tid < 32) {                                // prefetch k-block 0 into buf0
        tdm_load_tile64((unsigned)(size_t)(void*)(smem),            kG,  CH);
        tdm_load_tile64((unsigned)(size_t)(void*)(smem + TILE),     ktG, CHD);
        tdm_load_tile64((unsigned)(size_t)(void*)(smem + 2 * TILE), vtG, CS);
    }
#endif

    v8f oacc[4];
    #pragma unroll
    for (int t = 0; t < 4; ++t)
        #pragma unroll
        for (int r = 0; r < 8; ++r) oacc[t][r] = 0.f;
    float mrow[8], srow[8];
    #pragma unroll
    for (int r = 0; r < 8; ++r) { mrow[r] = -3.0e38f; srow[r] = 0.f; }

    // per-lane base pointer for the 537MB unif stream (offsets fold into insts)
    float* ubase = unif_out + (((size_t)b * CNH + h) * CS) * CS
                 + (size_t)(qbase + w16 + g * 8) * CS + l16;

    #pragma unroll 1
    for (int kb0 = 0; kb0 < CS; kb0 += 64) {
        __bf16* cur = smem + ((kb0 >> 6) & 1) * BUFE;
        __bf16* sK  = cur;                          // [k][d]
        __bf16* sKT = cur + TILE;                   // [k][e]
        __bf16* sVT = cur + 2 * TILE;               // [d][k]
        if (tid < 64) sMadd[tid] = (mask[b * CS + kb0 + tid] == 0) ? -1e30f : 0.0f;
#if USE_TDM
        if (tid < 32) {
            if (kb0 + 64 < CS) {                    // prefetch next block, then
                __bf16* nxt = smem + (((kb0 >> 6) + 1) & 1) * BUFE;
                tdm_load_tile64((unsigned)(size_t)(void*)(nxt),
                                kG + (size_t)(kb0 + 64) * CH, CH);
                tdm_load_tile64((unsigned)(size_t)(void*)(nxt + TILE),
                                ktG + (size_t)(kb0 + 64) * CHD, CHD);
                tdm_load_tile64((unsigned)(size_t)(void*)(nxt + 2 * TILE),
                                vtG + (kb0 + 64), CS);
                __builtin_amdgcn_s_wait_tensorcnt(3);  // current block's 3 done
            } else {
                __builtin_amdgcn_s_wait_tensorcnt(0);
            }
        }
        __syncthreads();
#else
        #pragma unroll
        for (int j = 0; j < 2; ++j) {
            int idx8 = tid + j * 256;
            int kk = idx8 >> 3, d0 = (idx8 & 7) * 8;
            *(v8bf*)&sK[kk * LD + d0]  = *(const v8bf*)&kG[(size_t)(kb0 + kk) * CH + d0];
            *(v8bf*)&sKT[kk * LD + d0] = *(const v8bf*)&ktG[(size_t)(kb0 + kk) * CHD + d0];
            *(v8bf*)&sVT[kk * LD + d0] = *(const v8bf*)&vtG[(size_t)kk * CS + kb0 + d0];
        }
        __syncthreads();
#endif

        float* up = ubase + kb0;
        float comb[4][8];
        #pragma unroll
        for (int nt = 0; nt < 4; ++nt) {
            v8f cv, cu;
            #pragma unroll
            for (int r = 0; r < 8; ++r) { cv[r] = 0.f; cu[r] = 0.f; }
            #pragma unroll
            for (int ds = 0; ds < 2; ++ds) {
                const __bf16* krow = &sK[(nt * 16 + l16) * LD + ds * 32];
                v16bf bk;
                ((v8bf*)&bk)[0] = *(const v8bf*)(krow + g * 8);
                ((v8bf*)&bk)[1] = *(const v8bf*)(krow + 16 + g * 8);
                cv = __builtin_amdgcn_wmma_f32_16x16x32_bf16(
                    false, qfrag[ds], false, bk, (short)0, cv, false, false);
                const __bf16* trow = &sKT[(nt * 16 + l16) * LD + ds * 32];
                v16bf bt;
                ((v8bf*)&bt)[0] = *(const v8bf*)(trow + g * 8);
                ((v8bf*)&bt)[1] = *(const v8bf*)(trow + 16 + g * 8);
                cu = __builtin_amdgcn_wmma_f32_16x16x32_bf16(
                    false, tfrag[ds], false, bt, (short)0, cu, false, false);
            }
            const float madd2 = sMadd[nt * 16 + l16];
            #pragma unroll
            for (int r = 0; r < 8; ++r) {
                float u = cu[r] * scale;
                __builtin_nontemporal_store(u, up + (size_t)r * CS + nt * 16);
                // t = e^-u = 2^(-u*log2e); sg = 1/(1+t); all raw HW transcendentals
                float t = __builtin_amdgcn_exp2f(cu[r] * (-sc2));
                float sg = __builtin_amdgcn_rcpf(1.0f + t);
                comb[nt][r] = cv[r] * sc2 + __builtin_amdgcn_logf(sg + 1e-6f) + madd2;
            }
        }
        // online softmax in log2 domain (rows live across 16 lanes of each half)
        float alpha[8];
        #pragma unroll
        for (int r = 0; r < 8; ++r) {
            float tmax = comb[0][r];
            #pragma unroll
            for (int nt = 1; nt < 4; ++nt) tmax = fmaxf(tmax, comb[nt][r]);
            #pragma unroll
            for (int mk = 1; mk < 16; mk <<= 1)
                tmax = fmaxf(tmax, __shfl_xor(tmax, mk, 32));
            float mnew = fmaxf(mrow[r], tmax);
            alpha[r] = __builtin_amdgcn_exp2f(mrow[r] - mnew);
            float ps = 0.f;
            #pragma unroll
            for (int nt = 0; nt < 4; ++nt) {
                float p = __builtin_amdgcn_exp2f(comb[nt][r] - mnew);
                comb[nt][r] = p;
                ps += p;
            }
            #pragma unroll
            for (int mk = 1; mk < 16; mk <<= 1) ps += __shfl_xor(ps, mk, 32);
            srow[r] = srow[r] * alpha[r] + ps;
            mrow[r] = mnew;
        }
        #pragma unroll
        for (int t = 0; t < 4; ++t)
            #pragma unroll
            for (int r = 0; r < 8; ++r) oacc[t][r] *= alpha[r];
        // P (C layout) -> per-wave LDS -> A fragments
        #pragma unroll
        for (int nt = 0; nt < 4; ++nt)
            #pragma unroll
            for (int r = 0; r < 8; ++r)
                myP[(g * 8 + r) * LD + nt * 16 + l16] = f2bf(comb[nt][r]);
        #pragma unroll
        for (int nt = 0; nt < 4; ++nt) {
            #pragma unroll
            for (int ks = 0; ks < 2; ++ks) {
                const __bf16* prow = &myP[l16 * LD + ks * 32];
                v16bf ap;
                ((v8bf*)&ap)[0] = *(const v8bf*)(prow + g * 8);
                ((v8bf*)&ap)[1] = *(const v8bf*)(prow + 16 + g * 8);
                const __bf16* vrow = &sVT[(nt * 16 + l16) * LD + ks * 32];
                v16bf bv;
                ((v8bf*)&bv)[0] = *(const v8bf*)(vrow + g * 8);
                ((v8bf*)&bv)[1] = *(const v8bf*)(vrow + 16 + g * 8);
                oacc[nt] = __builtin_amdgcn_wmma_f32_16x16x32_bf16(
                    false, ap, false, bv, (short)0, oacc[nt], false, false);
            }
        }
        __syncthreads();                           // tiles reusable next iter
    }
    float* op = aout + ((size_t)(b * CS + qbase + w16 + g * 8)) * CH + h * CHD + l16;
    #pragma unroll
    for (int r = 0; r < 8; ++r) {
        float inv = __builtin_amdgcn_rcpf(srow[r]);
        #pragma unroll
        for (int nt = 0; nt < 4; ++nt)
            op[r * CH + nt * 16] = oacc[nt][r] * inv;
    }
}

// ---------------------------------------------------------------------------
extern "C" void kernel_launch(void* const* d_in, const int* in_sizes, int n_in,
                              void* d_out, int out_size, void* d_ws, size_t ws_size,
                              hipStream_t stream) {
    (void)in_sizes; (void)n_in; (void)out_size; (void)ws_size;
    const float* x    = (const float*)d_in[0];
    const int*   mask = (const int*)d_in[1];
    const float* Wq   = (const float*)d_in[2];
    const float* bq   = (const float*)d_in[3];
    const float* Wk   = (const float*)d_in[4];
    const float* bk   = (const float*)d_in[5];
    const float* Wv   = (const float*)d_in[6];
    const float* bv   = (const float*)d_in[7];
    const float* Wo   = (const float*)d_in[8];
    const float* bo   = (const float*)d_in[9];
    const float* Wti  = (const float*)d_in[10];
    const float* bti  = (const float*)d_in[11];
    const float* temb = (const float*)d_in[12];
    const float* Wqt  = (const float*)d_in[13];
    const float* bqt  = (const float*)d_in[14];
    const float* Wkt  = (const float*)d_in[15];
    const float* bkt  = (const float*)d_in[16];
    const float* Wbil = (const float*)d_in[17];

    constexpr size_t NBSH = (size_t)CB * CS * CH;              // 4M elems
    constexpr size_t NHH  = (size_t)CH * CH;                   // 1M elems
    constexpr size_t OFF_XB    = 0;
    constexpr size_t OFF_WQB   = OFF_XB    + NBSH * 2;
    constexpr size_t OFF_WKB   = OFF_WQB   + NHH * 2;
    constexpr size_t OFF_WVB   = OFF_WKB   + NHH * 2;
    constexpr size_t OFF_WOB   = OFF_WVB   + NHH * 2;
    constexpr size_t OFF_QB    = OFF_WOB   + NHH * 2;
    constexpr size_t OFF_KB    = OFF_QB    + NBSH * 2;
    constexpr size_t OFF_VB    = OFF_KB    + NBSH * 2;
    constexpr size_t OFF_KTB   = OFF_VB    + NBSH * 2;
    constexpr size_t OFF_QTF   = OFF_KTB   + (size_t)CB * CS * CHD * 2;
    constexpr size_t OFF_QTBB  = OFF_QTF   + (size_t)CB * CS * CHD * 4;
    constexpr size_t OFF_AOUT  = OFF_QTBB  + (size_t)CB * CNH * CS * CHD * 2;
    constexpr size_t OFF_AOUTB = OFF_AOUT  + NBSH * 4;
    constexpr size_t OFF_VTB   = OFF_AOUTB + NBSH * 2;

    char* ws = (char*)d_ws;
    __bf16* xb    = (__bf16*)(ws + OFF_XB);
    __bf16* wqb   = (__bf16*)(ws + OFF_WQB);
    __bf16* wkb   = (__bf16*)(ws + OFF_WKB);
    __bf16* wvb   = (__bf16*)(ws + OFF_WVB);
    __bf16* wob   = (__bf16*)(ws + OFF_WOB);
    __bf16* qbuf  = (__bf16*)(ws + OFF_QB);
    __bf16* kbuf  = (__bf16*)(ws + OFF_KB);
    __bf16* vbuf  = (__bf16*)(ws + OFF_VB);
    __bf16* ktbuf = (__bf16*)(ws + OFF_KTB);
    float*  qtf   = (float*) (ws + OFF_QTF);
    __bf16* qtbb  = (__bf16*)(ws + OFF_QTBB);
    float*  aout  = (float*) (ws + OFF_AOUT);
    __bf16* aoutb = (__bf16*)(ws + OFF_AOUTB);
    __bf16* vtb   = (__bf16*)(ws + OFF_VTB);

    float* out0 = (float*)d_out;                               // [B,S,H]
    float* out1 = out0 + NBSH;                                 // type_logits [B,S,NT]
    float* out2 = out1 + (size_t)CB * CS * CNT;                // unif_logits [B,NH,S,S]

    const int M = CB * CS;

    cvt_kernel<<<1024, 256, 0, stream>>>(x,  xb,  (int)NBSH);
    cvt_kernel<<<512,  256, 0, stream>>>(Wq, wqb, (int)NHH);
    cvt_kernel<<<512,  256, 0, stream>>>(Wk, wkb, (int)NHH);
    cvt_kernel<<<512,  256, 0, stream>>>(Wv, wvb, (int)NHH);
    cvt_kernel<<<512,  256, 0, stream>>>(Wo, wob, (int)NHH);

    type_kernel<<<M, 128, 0, stream>>>(x, Wti, bti, temb, Wqt, bqt, Wkt, bkt,
                                       out1, qtf, ktbuf);
    qtbil_kernel<<<dim3(M, CNH), CHD, 0, stream>>>(qtf, Wbil, qtbb);

    dim3 ggrid(CH / 128, M / 128);
    gemm_bf16<true><<<ggrid, 256, 0, stream>>>(xb, wqb, bq, qbuf, M, CH, CH);
    gemm_bf16<true><<<ggrid, 256, 0, stream>>>(xb, wkb, bk, kbuf, M, CH, CH);
    gemm_bf16<true><<<ggrid, 256, 0, stream>>>(xb, wvb, bv, vbuf, M, CH, CH);

    vtrans_kernel<<<dim3(CS / 64, CNH, CB), 256, 0, stream>>>(vbuf, vtb);

    flash_kernel<<<dim3(CS / 128, CNH, CB), 256, 0, stream>>>(
        qbuf, kbuf, vtb, qtbb, ktbuf, mask, out2, aout);

    cvt_kernel<<<1024, 256, 0, stream>>>(aout, aoutb, (int)NBSH);
    gemm_bf16<false><<<ggrid, 256, 0, stream>>>(aoutb, wob, bo, out0, M, CH, CH);
}
